// GraphConvlayer_23587960389875
// MI455X (gfx1250) — matrix operand order repaired and checked by reading, not compiled, not run on hardware
//
#include <hip/hip_runtime.h>
#include <hip/hip_bf16.h>
#include <math.h>

#define N_NODES 50000
#define N_EDGES 400000
#define DIM     512
#define KTILES  16                    // 512 / 32
#define MTILE   32                    // rows per workgroup
#define WT_ROW  40                    // padded bf16 row: 80B = 16B-aligned, 20-dword bank stride
#define WT_TILE_BYTES (DIM * WT_ROW * 2)  // 40960B per K-step tile

typedef __attribute__((ext_vector_type(16))) __bf16 v16bf;
typedef __attribute__((ext_vector_type(8)))  float  v8f;
typedef __attribute__((address_space(3))) void lds_void;

__device__ __forceinline__ __bf16 f2bf(float f) {
    union { float f; unsigned u; } in;
    in.f = f;
    unsigned u = in.u + 0x7FFFu + ((in.u >> 16) & 1u);  // round-to-nearest-even
    union { unsigned short s; __bf16 b; } out;
    out.s = (unsigned short)(u >> 16);
    return out.b;
}

// CDNA5 async global->LDS copy (16B per lane), tracked by ASYNCcnt.
__device__ __forceinline__ void async_cp16(void* lds, const void* g) {
    unsigned loff = (unsigned)(uintptr_t)(lds_void*)lds;      // 32-bit LDS offset
    asm volatile("global_load_async_to_lds_b128 %0, %1, off"
                 :: "v"(loff), "v"((unsigned long long)(uintptr_t)g)
                 : "memory");
}

__device__ __forceinline__ void wait_async0() {
#if __has_builtin(__builtin_amdgcn_s_wait_asynccnt)
    __builtin_amdgcn_s_wait_asynccnt(0);
#else
    asm volatile("s_wait_asynccnt 0x0" ::: "memory");
#endif
}

// ---------------- Kernel 1: degree counting ----------------
__global__ void deg_kernel(const long long* __restrict__ src,
                           const long long* __restrict__ dst,
                           float* __restrict__ deg_out,
                           float* __restrict__ deg_in) {
    int e = blockIdx.x * blockDim.x + threadIdx.x;
    if (e < N_EDGES) {
        unsafeAtomicAdd(&deg_out[(int)src[e]], 1.0f);
        unsafeAtomicAdd(&deg_in[(int)dst[e]], 1.0f);
    }
}

// ---------------- Kernel 2: SpMM scatter-add (one wave per edge) ----------------
__global__ __launch_bounds__(256) void spmm_kernel(const float* __restrict__ x,
                                                   const long long* __restrict__ src,
                                                   const long long* __restrict__ dst,
                                                   const float* __restrict__ deg_out,
                                                   float* __restrict__ agg) {
    int wave = threadIdx.x >> 5;
    int lane = threadIdx.x & 31;
    int e = blockIdx.x * 8 + wave;
    if (e >= N_EDGES) return;
    int s = (int)src[e];
    int d = (int)dst[e];
    float sc = rsqrtf(fmaxf(deg_out[s], 1.0f));
    const float* xs = x + (long long)s * DIM;
    float* ad = agg + (long long)d * DIM;
#pragma unroll
    for (int i = 0; i < 16; ++i) {
        int c = lane + 32 * i;
        unsafeAtomicAdd(&ad[c], xs[c] * sc);  // agg (102MB) is L2-resident -> on-chip atomics
    }
}

// ---------------- Kernel 3a: W -> bf16, transposed K-step tiles matching LDS image ----------------
// wbf layout: tile kt (40960B) = bf16[512 n][40 kk'], cols 0..31 = W[kt*32+kk][n], 32..39 = 0 pad.
__global__ __launch_bounds__(256) void wconv_kernel(const float* __restrict__ Wm,
                                                    __bf16* __restrict__ wbf) {
    int idx = blockIdx.x * 256 + threadIdx.x;   // 16*512*10 = 81920 groups of 4 cols
    int n  = idx & 511;
    int t2 = idx >> 9;
    int g4 = t2 % 10;
    int kt = t2 / 10;
    __bf16* dp = wbf + (size_t)kt * (DIM * WT_ROW) + n * WT_ROW + g4 * 4;
    if (g4 < 8) {
        int k = kt * 32 + g4 * 4;
        const float* wp = Wm + (long long)k * DIM + n;   // coalesced over n
        dp[0] = f2bf(wp[0]);
        dp[1] = f2bf(wp[DIM]);
        dp[2] = f2bf(wp[2 * DIM]);
        dp[3] = f2bf(wp[3 * DIM]);
    } else {                                              // zero padding cols 32..39
        dp[0] = f2bf(0.f); dp[1] = f2bf(0.f); dp[2] = f2bf(0.f); dp[3] = f2bf(0.f);
    }
}

// ---------------- Kernel 3b: agg * rsqrt(deg_in) -> bf16 ----------------
__global__ __launch_bounds__(256) void aconv_kernel(const float* __restrict__ agg,
                                                    const float* __restrict__ deg_in,
                                                    __bf16* __restrict__ abf) {
    long long idx = (long long)blockIdx.x * 256 + threadIdx.x;  // 6.4M float4 groups
    int r  = (int)(idx >> 7);
    int c4 = ((int)idx & 127) * 4;
    float s = rsqrtf(fmaxf(deg_in[r], 1.0f));
    const float4 a4 = *(const float4*)(agg + (long long)r * DIM + c4);
    __bf16* dp = abf + (long long)r * DIM + c4;
    dp[0] = f2bf(a4.x * s); dp[1] = f2bf(a4.y * s);
    dp[2] = f2bf(a4.z * s); dp[3] = f2bf(a4.w * s);
}

// ---------------- Kernel 4: bf16-WMMA GEMM (async-LDS staged, 32-row M-tile) + bias + LN + GELU ----------------
__global__ __launch_bounds__(256) void gemm_ln_gelu(const __bf16* __restrict__ abf,
                                                    const __bf16* __restrict__ wbf,
                                                    const float* __restrict__ bias,
                                                    const float* __restrict__ gamma,
                                                    const float* __restrict__ beta,
                                                    float* __restrict__ out) {
    __shared__ union {
        struct {
            __bf16 At[MTILE][WT_ROW]; // 2560B, rows 80B: 16B-aligned, bank-clean
            __bf16 Wt[DIM][WT_ROW];   // 40960B, same row geometry as global tile
        } g;
        float stage[16][520];         // fp32 epilogue staging (two passes of 16 rows)
    } sm __attribute__((aligned(16)));

    const int tid  = threadIdx.x;
    const int lane = tid & 31;
    const int w    = tid >> 5;     // wave id 0..7
    const int hf   = lane >> 4;    // half-wave selector
    const int nql  = lane & 15;
    const int r0   = blockIdx.x * MTILE;

    v8f acc[2][4];                 // [m-group][n-subtile]
#pragma unroll
    for (int mg = 0; mg < 2; ++mg)
#pragma unroll
        for (int j = 0; j < 4; ++j) acc[mg][j] = (v8f){};

    const char* abase = (const char*)(abf + (long long)r0 * DIM);

    for (int kt = 0; kt < KTILES; ++kt) {
        // --- async-stage W tile: global layout == LDS image -> straight 40KB copy ---
        const char* wt_g = (const char*)wbf + (size_t)kt * WT_TILE_BYTES;
        char* wt_l = (char*)&sm.g.Wt[0][0];
#pragma unroll
        for (int j = 0; j < 10; ++j) {
            int c = (tid + 256 * j) * 16;
            async_cp16(wt_l + c, wt_g + c);
        }
        // --- async-stage A tile: 32 rows x 64B (rows padded to 80B in LDS) ---
        if (tid < 128) {
            int m = tid >> 2, seg = tid & 3;
            async_cp16((char*)&sm.g.At[m][0] + seg * 16,
                       abase + (long long)m * (DIM * 2) + kt * 64 + seg * 16);
        }
        wait_async0();
        __syncthreads();

        // --- A fragments, one per 16-row m-group (ISA 16-bit A 16x32 layout) ---
        union { v16bf v; uint4 q[2]; } fa[2];
#pragma unroll
        for (int mg = 0; mg < 2; ++mg) {
            fa[mg].q[0] = *(const uint4*)&sm.g.At[mg * 16 + nql][hf * 8];
            fa[mg].q[1] = *(const uint4*)&sm.g.At[mg * 16 + nql][16 + hf * 8];
        }

        // --- 4 B fragments; each feeds two WMMAs (both m-groups) ---
#pragma unroll
        for (int j = 0; j < 4; ++j) {
            int ng = (w * 4 + j) * 16 + nql;
            union { v16bf v; uint4 q[2]; } fb;
            const __bf16* bp = &sm.g.Wt[ng][hf * 8];
            fb.q[0] = *(const uint4*)bp;
            fb.q[1] = *(const uint4*)(bp + 16);
            acc[0][j] = __builtin_amdgcn_wmma_f32_16x16x32_bf16(
                false, fa[0].v, false, fb.v, (short)0, acc[0][j], false, false);
            acc[1][j] = __builtin_amdgcn_wmma_f32_16x16x32_bf16(
                false, fa[1].v, false, fb.v, (short)0, acc[1][j], false, false);
        }
        __syncthreads();
    }

    // --- epilogue: two passes of 16 rows through the fp32 staging buffer ---
#pragma unroll
    for (int mg = 0; mg < 2; ++mg) {
        // stage accumulators (C layout: VGPR r -> local M=r+8*hf, N=nql)
#pragma unroll
        for (int j = 0; j < 4; ++j) {
            int nbase = (w * 4 + j) * 16 + nql;
#pragma unroll
            for (int r = 0; r < 8; ++r) {
                sm.stage[r + 8 * hf][nbase] = acc[mg][j][r];
            }
        }
        __syncthreads();

        // LayerNorm + exact GELU: each wave owns 2 rows of this pass
#pragma unroll
        for (int rr = 0; rr < 2; ++rr) {
            int m  = w * 2 + rr;
            int gr = r0 + mg * 16 + m;
            float vals[16];
            float sum = 0.f, sq = 0.f;
#pragma unroll
            for (int i = 0; i < 16; ++i) {
                int c = lane + 32 * i;
                float v = sm.stage[m][c] + bias[c];
                vals[i] = v;
                sum += v;
                sq  += v * v;
            }
#pragma unroll
            for (int off = 16; off > 0; off >>= 1) {
                sum += __shfl_xor(sum, off);
                sq  += __shfl_xor(sq, off);
            }
            float mu   = sum * (1.0f / 512.0f);
            float var  = sq * (1.0f / 512.0f) - mu * mu;
            float rstd = rsqrtf(var + 1e-5f);
            if (gr < N_NODES) {
                float* op = out + (long long)gr * DIM;
#pragma unroll
                for (int i = 0; i < 16; ++i) {
                    int c = lane + 32 * i;
                    float y = (vals[i] - mu) * rstd * gamma[c] + beta[c];
                    op[c] = 0.5f * y * (1.0f + erff(y * 0.70710678118654752440f));
                }
            }
        }
        __syncthreads();   // staging buffer free for next pass
    }
}

extern "C" void kernel_launch(void* const* d_in, const int* in_sizes, int n_in,
                              void* d_out, int out_size, void* d_ws, size_t ws_size,
                              hipStream_t stream) {
    const float*     x     = (const float*)d_in[0];
    const float*     Wm    = (const float*)d_in[1];
    const float*     b     = (const float*)d_in[2];
    const float*     gamma = (const float*)d_in[3];
    const float*     beta  = (const float*)d_in[4];
    const long long* src   = (const long long*)d_in[5];
    const long long* dst   = (const long long*)d_in[6];
    float* out = (float*)d_out;

    // workspace layout
    char* ws = (char*)d_ws;
    float*  deg_out = (float*)(ws);                          // 200KB
    float*  deg_in  = (float*)(ws + (256u << 10));           // 200KB
    __bf16* wbf     = (__bf16*)(ws + (512u << 10));          // 640KB (16 x 40KB tiles)
    __bf16* abf     = (__bf16*)(ws + (2u << 20));            // 51.2MB
    float*  agg     = (float*)(ws + (56ull << 20));          // 102.4MB

    hipMemsetAsync(ws, 0, (512u << 10), stream);                              // degrees
    hipMemsetAsync(agg, 0, (size_t)N_NODES * DIM * sizeof(float), stream);    // agg

    deg_kernel<<<(N_EDGES + 255) / 256, 256, 0, stream>>>(src, dst, deg_out, deg_in);
    spmm_kernel<<<N_EDGES / 8, 256, 0, stream>>>(x, src, dst, deg_out, agg);
    wconv_kernel<<<(KTILES * DIM * 10) / 256, 256, 0, stream>>>(Wm, wbf);
    aconv_kernel<<<(N_NODES * DIM / 4) / 256, 256, 0, stream>>>(agg, deg_in, abf);
    gemm_ln_gelu<<<(N_NODES + MTILE - 1) / MTILE, 256, 0, stream>>>(abf, wbf, b, gamma, beta, out);
}
